// MultiHeadSelfAttention_30863634989797
// MI455X (gfx1250) — compile-verified, matrix operand order checked by reading, not run on hardware
//
#include <hip/hip_runtime.h>
#include <hip/hip_bf16.h>
#include <stdint.h>

// ---------------------------------------------------------------------------
// MHA forward for MI455X (gfx1250, wave32, WMMA).
//   B=4, S=1024, D=1024, H=16, Dh=64.
// Math path: bf16 V_WMMA_F32_16X16X32_BF16, fp32 accumulate.
// fp32->bf16 conversion hoisted into streaming kernels (bandwidth-bound, ALU
// free there); GEMM loops stage BOTH operands with GLOBAL_LOAD_ASYNC_TO_LDS
// (ASYNCcnt) into double-buffered LDS -> pure async-DMA + WMMA steady state.
// ---------------------------------------------------------------------------

typedef __attribute__((ext_vector_type(16))) __bf16 v16bf;
typedef __attribute__((ext_vector_type(8)))  float  v8f;
typedef __attribute__((ext_vector_type(4)))  int    v4i;

#define WMMA_BF16(a, b, c) \
    __builtin_amdgcn_wmma_f32_16x16x32_bf16(false, (a), false, (b), (short)0, (c), false, false)

#if defined(__gfx1250__) && __has_builtin(__builtin_amdgcn_global_load_async_to_lds_b128)
#define HAVE_ASYNC_LDS 1
#endif

static __device__ __forceinline__ uint16_t f32_to_bf16(float f) {
    uint32_t u = __float_as_uint(f);
    u += 0x7FFFu + ((u >> 16) & 1u);   // round-to-nearest-even
    return (uint16_t)(u >> 16);
}

static __device__ __forceinline__ uint32_t pack2_bf16(float x, float y) {
    return (uint32_t)f32_to_bf16(x) | ((uint32_t)f32_to_bf16(y) << 16);
}

#ifdef HAVE_ASYNC_LDS
// Builtin signature (probe-confirmed via diagnostic): v4i __global* src,
// v4i __shared* dst, imm offset, imm cpol.
typedef __attribute__((address_space(1))) v4i* v4i_gp;   // global (AS1)
typedef __attribute__((address_space(3))) v4i* v4i_lp;   // LDS (AS3, 32-bit)
static __device__ __forceinline__ void async_b128(const void* gp, const void* lp) {
    __builtin_amdgcn_global_load_async_to_lds_b128(
        (v4i_gp)(uintptr_t)gp,
        (v4i_lp)(uint32_t)(uintptr_t)lp,   // low 32 bits of generic = LDS offset
        0, 0);
}
static __device__ __forceinline__ void wait_async0() {
#if __has_builtin(__builtin_amdgcn_s_wait_asynccnt)
    __builtin_amdgcn_s_wait_asynccnt(0);
#else
    asm volatile("s_wait_asynccnt 0x0" ::: "memory");
#endif
}
#endif

// ---------------------------------------------------------------------------
// Kernel 0: streaming fp32 -> bf16 conversion (bandwidth-bound).
// n4 = element count in float4 units.
// ---------------------------------------------------------------------------
__global__ __launch_bounds__(256) void cvt_bf16_kernel(
    const float* __restrict__ src, uint16_t* __restrict__ dst, int n4)
{
    const int i = blockIdx.x * 256 + threadIdx.x;
    if (i < n4) {
        const float4 f = ((const float4*)src)[i];
        uint2 w;
        w.x = pack2_bf16(f.x, f.y);
        w.y = pack2_bf16(f.z, f.w);
        ((uint2*)dst)[i] = w;
    }
}

// ---------------------------------------------------------------------------
// Shared GEMM tile machinery: 128x64 block tile, 4 waves, 32x64 per wave,
// BK=32, double-buffered LDS, async-DMA staging of bf16 A and B tiles.
//   A tile: 128 rows x 32 cols (thread t -> row t, 4x b128)
//   B tile:  64 rows x 32 cols (thread t -> row t>>1, half t&1, 2x b128)
// ---------------------------------------------------------------------------

// ---------------------------------------------------------------------------
// Kernel 1: qkv = Xb @ Wqkvb^T  (M=4096, K=1024, N=3072), fused RoPE epilogue.
// Writes bf16: q,k as [B*H, S, Dh]; v transposed as [B*H, Dh, S].
// ---------------------------------------------------------------------------
__global__ __launch_bounds__(128) void qkv_rope_kernel(
    const uint16_t* __restrict__ Xb, const uint16_t* __restrict__ Wb,
    const int* __restrict__ pos,
    uint16_t* __restrict__ qb, uint16_t* __restrict__ kb, uint16_t* __restrict__ vT)
{
    __shared__ alignas(32) uint16_t sA[2][128 * 32];
    __shared__ alignas(32) uint16_t sB[2][64 * 32];

    const int n0 = blockIdx.x * 64;    // 48 blocks -> N=3072
    const int m0 = blockIdx.y * 128;   // 32 blocks -> M=4096
    const int t = threadIdx.x, wave = t >> 5, lane = t & 31;
    const int ln = lane & 15, hi = lane >> 4;

    const uint16_t* Arow = Xb + (size_t)(m0 + t) * 1024;
    const uint16_t* Brow = Wb + (size_t)(n0 + (t >> 1)) * 1024 + (t & 1) * 16;

    v8f acc[2][4] = {};
#ifndef HAVE_ASYNC_LDS
    uint4 au[4], bu[2];
#endif

    auto load_tiles = [&](int k0, int buf) {
#ifdef HAVE_ASYNC_LDS
        #pragma unroll
        for (int i = 0; i < 4; i++)
            async_b128(Arow + k0 + i * 8, &sA[buf][t * 32 + i * 8]);
        #pragma unroll
        for (int i = 0; i < 2; i++)
            async_b128(Brow + k0 + i * 8, &sB[buf][(t >> 1) * 32 + (t & 1) * 16 + i * 8]);
#else
        #pragma unroll
        for (int i = 0; i < 4; i++) au[i] = *(const uint4*)(Arow + k0 + i * 8);
        #pragma unroll
        for (int i = 0; i < 2; i++) bu[i] = *(const uint4*)(Brow + k0 + i * 8);
        (void)buf;
#endif
    };
    auto store_tiles = [&](int buf) {
#ifndef HAVE_ASYNC_LDS
        #pragma unroll
        for (int i = 0; i < 4; i++)
            *(uint4*)&sA[buf][t * 32 + i * 8] = au[i];
        #pragma unroll
        for (int i = 0; i < 2; i++)
            *(uint4*)&sB[buf][(t >> 1) * 32 + (t & 1) * 16 + i * 8] = bu[i];
#else
        (void)buf;
#endif
    };

    load_tiles(0, 0);
    store_tiles(0);
#ifdef HAVE_ASYNC_LDS
    wait_async0();
#endif
    __syncthreads();

    for (int k0 = 0; k0 < 1024; k0 += 32) {
        const int cur = (k0 >> 5) & 1;
        const bool more = (k0 + 32) < 1024;
        if (more) load_tiles(k0 + 32, cur ^ 1);   // async DMA of next tile

        const v16bf a0 = *(const v16bf*)&sA[cur][(wave * 32 + ln) * 32 + hi * 16];
        const v16bf a1 = *(const v16bf*)&sA[cur][(wave * 32 + 16 + ln) * 32 + hi * 16];
        const v16bf b0 = *(const v16bf*)&sB[cur][(0 * 16 + ln) * 32 + hi * 16];
        const v16bf b1 = *(const v16bf*)&sB[cur][(1 * 16 + ln) * 32 + hi * 16];
        const v16bf b2 = *(const v16bf*)&sB[cur][(2 * 16 + ln) * 32 + hi * 16];
        const v16bf b3 = *(const v16bf*)&sB[cur][(3 * 16 + ln) * 32 + hi * 16];
        acc[0][0] = WMMA_BF16(a0, b0, acc[0][0]);
        acc[1][0] = WMMA_BF16(a1, b0, acc[1][0]);
        acc[0][1] = WMMA_BF16(a0, b1, acc[0][1]);
        acc[1][1] = WMMA_BF16(a1, b1, acc[1][1]);
        acc[0][2] = WMMA_BF16(a0, b2, acc[0][2]);
        acc[1][2] = WMMA_BF16(a1, b2, acc[1][2]);
        acc[0][3] = WMMA_BF16(a0, b3, acc[0][3]);
        acc[1][3] = WMMA_BF16(a1, b3, acc[1][3]);

        if (more) store_tiles(cur ^ 1);
#ifdef HAVE_ASYNC_LDS
        if (more) wait_async0();
#endif
        __syncthreads();
    }

    // Epilogue: RoPE on q/k segments, scatter to attention-friendly layouts.
    const float LN1E4 = 9.210340371976184f;   // ln(10000)
    #pragma unroll
    for (int mt = 0; mt < 2; mt++) {
        #pragma unroll
        for (int nt = 0; nt < 4; nt++) {
            const int col = n0 + nt * 16 + ln;   // uniform segment per tile
            const int which = col >> 10;         // 0=q 1=k 2=v
            const int cd = col & 1023;
            const int h = cd >> 6, d = cd & 63;
            #pragma unroll
            for (int r = 0; r < 8; r++) {
                const int m = m0 + wave * 32 + mt * 16 + r + hi * 8;
                const int bidx = m >> 10, s = m & 1023;
                float val = acc[mt][nt][r];
                const float partner = __shfl_xor(val, 1, 32);  // adjacent (d^1) pair
                if (which < 2) {
                    int p = pos[s]; p = p < 0 ? 0 : (p > 4095 ? 4095 : p);
                    const float invf = __expf(-((float)(d & ~1) / 64.0f) * LN1E4);
                    const float ang  = (float)p * invf;
                    const float rot  = (d & 1) ? partner : -partner;
                    val = val * __cosf(ang) + rot * __sinf(ang);
                }
                const uint16_t hv = f32_to_bf16(val);
                const size_t bh = (size_t)(bidx * 16 + h);
                if (which == 0)      qb[(bh * 1024 + s) * 64 + d] = hv;
                else if (which == 1) kb[(bh * 1024 + s) * 64 + d] = hv;
                else                 vT[(bh * 64 + d) * 1024 + s] = hv;
            }
        }
    }
}

// ---------------------------------------------------------------------------
// Kernel 2: flash-style causal attention. 1 wave per (b,h, 16-row q-tile).
// Single-wave workgroup => barriers are free (S_NOP per ISA).
// ---------------------------------------------------------------------------
__global__ __launch_bounds__(32) void attn_kernel(
    const uint16_t* __restrict__ qb, const uint16_t* __restrict__ kb,
    const uint16_t* __restrict__ vT, uint16_t* __restrict__ ob)
{
    __shared__ alignas(32) uint16_t sP[16 * 32];

    const int qt = blockIdx.x & 63;      // q tile within head (S/16 = 64)
    const int bh = blockIdx.x >> 6;      // 0..63 (= b*16 + h)
    const int lane = threadIdx.x, ln = lane & 15, hi = lane >> 4;

    const uint16_t* qp = qb + (size_t)bh * 1024 * 64;
    const uint16_t* kp = kb + (size_t)bh * 1024 * 64;
    const uint16_t* vp = vT + (size_t)bh * 64 * 1024;

    const v16bf aq0 = *(const v16bf*)(qp + (qt * 16 + ln) * 64 + hi * 16);
    const v16bf aq1 = *(const v16bf*)(qp + (qt * 16 + ln) * 64 + 32 + hi * 16);

    v8f o[4] = {};
    float mr[8], lr[8];
    #pragma unroll
    for (int r = 0; r < 8; r++) { mr[r] = -1e30f; lr[r] = 0.0f; }

    const int kend = qt * 16 + 16;       // causal: keys < kend
    for (int j0 = 0; j0 < kend; j0 += 32) {
        // ---- hoisted K fragments, then 4 back-to-back score WMMAs ----
        const uint16_t* k0p = kp + (size_t)(j0 + ln) * 64;
        const uint16_t* k1p = kp + (size_t)(j0 + 16 + ln) * 64;
        const v16bf bk00 = *(const v16bf*)(k0p + hi * 16);
        const v16bf bk01 = *(const v16bf*)(k0p + 32 + hi * 16);
        const v16bf bk10 = *(const v16bf*)(k1p + hi * 16);
        const v16bf bk11 = *(const v16bf*)(k1p + 32 + hi * 16);
        v8f s0 = {}, s1 = {};
        s0 = WMMA_BF16(aq0, bk00, s0);
        s1 = WMMA_BF16(aq0, bk10, s1);
        s0 = WMMA_BF16(aq1, bk01, s0);
        s1 = WMMA_BF16(aq1, bk11, s1);

        // ---- causal mask + online softmax (row reductions in 16-lane half) ----
        float fac[8];
        #pragma unroll
        for (int r = 0; r < 8; r++) {
            const int qrow = qt * 16 + r + hi * 8;
            float x0 = s0[r] * 0.125f; if (j0 + ln      > qrow) x0 = -1e30f;
            float x1 = s1[r] * 0.125f; if (j0 + 16 + ln > qrow) x1 = -1e30f;
            float mx = fmaxf(x0, x1);
            mx = fmaxf(mx, __shfl_xor(mx, 1, 32));
            mx = fmaxf(mx, __shfl_xor(mx, 2, 32));
            mx = fmaxf(mx, __shfl_xor(mx, 4, 32));
            mx = fmaxf(mx, __shfl_xor(mx, 8, 32));
            const float nm = fmaxf(mr[r], mx);
            const float p0 = __expf(x0 - nm), p1 = __expf(x1 - nm);
            float ps = p0 + p1;
            ps += __shfl_xor(ps, 1, 32); ps += __shfl_xor(ps, 2, 32);
            ps += __shfl_xor(ps, 4, 32); ps += __shfl_xor(ps, 8, 32);
            fac[r] = __expf(mr[r] - nm);
            lr[r]  = lr[r] * fac[r] + ps;
            mr[r]  = nm;
            sP[(r + hi * 8) * 32 + ln]      = f32_to_bf16(p0);   // C-layout ->
            sP[(r + hi * 8) * 32 + 16 + ln] = f32_to_bf16(p1);   // A-layout bounce
        }
        __syncthreads();

        // ---- O += P @ V (V transposed => contiguous B-operand) ----
        const v16bf apv = *(const v16bf*)&sP[ln * 32 + hi * 16];
        const v16bf bv0 = *(const v16bf*)(vp + (size_t)(0 * 16 + ln) * 1024 + j0 + hi * 16);
        const v16bf bv1 = *(const v16bf*)(vp + (size_t)(1 * 16 + ln) * 1024 + j0 + hi * 16);
        const v16bf bv2 = *(const v16bf*)(vp + (size_t)(2 * 16 + ln) * 1024 + j0 + hi * 16);
        const v16bf bv3 = *(const v16bf*)(vp + (size_t)(3 * 16 + ln) * 1024 + j0 + hi * 16);
        #pragma unroll
        for (int c = 0; c < 4; c++) {
            #pragma unroll
            for (int r = 0; r < 8; r++) o[c][r] *= fac[r];
        }
        o[0] = WMMA_BF16(apv, bv0, o[0]);
        o[1] = WMMA_BF16(apv, bv1, o[1]);
        o[2] = WMMA_BF16(apv, bv2, o[2]);
        o[3] = WMMA_BF16(apv, bv3, o[3]);
        __syncthreads();
    }

    const int b_ = bh >> 4, h = bh & 15;
    #pragma unroll
    for (int c = 0; c < 4; c++) {
        #pragma unroll
        for (int r = 0; r < 8; r++) {
            const int qrow = qt * 16 + r + hi * 8;
            const float v = o[c][r] / lr[r];
            ob[((size_t)(b_ * 1024 + qrow)) * 1024 + h * 64 + c * 16 + ln] = f32_to_bf16(v);
        }
    }
}

// ---------------------------------------------------------------------------
// Kernel 3: out = attn @ Woutb^T  (M=4096, K=1024, N=1024), fp32 output.
// Same async-staged bf16 GEMM structure as kernel 1.
// ---------------------------------------------------------------------------
__global__ __launch_bounds__(128) void out_gemm_kernel(
    const uint16_t* __restrict__ A, const uint16_t* __restrict__ Wb,
    float* __restrict__ out)
{
    __shared__ alignas(32) uint16_t sA[2][128 * 32];
    __shared__ alignas(32) uint16_t sB[2][64 * 32];

    const int n0 = blockIdx.x * 64;    // 16 blocks
    const int m0 = blockIdx.y * 128;   // 32 blocks
    const int t = threadIdx.x, wave = t >> 5, lane = t & 31;
    const int ln = lane & 15, hi = lane >> 4;

    const uint16_t* Arow = A  + (size_t)(m0 + t) * 1024;
    const uint16_t* Brow = Wb + (size_t)(n0 + (t >> 1)) * 1024 + (t & 1) * 16;

    v8f acc[2][4] = {};
#ifndef HAVE_ASYNC_LDS
    uint4 au[4], bu[2];
#endif

    auto load_tiles = [&](int k0, int buf) {
#ifdef HAVE_ASYNC_LDS
        #pragma unroll
        for (int i = 0; i < 4; i++)
            async_b128(Arow + k0 + i * 8, &sA[buf][t * 32 + i * 8]);
        #pragma unroll
        for (int i = 0; i < 2; i++)
            async_b128(Brow + k0 + i * 8, &sB[buf][(t >> 1) * 32 + (t & 1) * 16 + i * 8]);
#else
        #pragma unroll
        for (int i = 0; i < 4; i++) au[i] = *(const uint4*)(Arow + k0 + i * 8);
        #pragma unroll
        for (int i = 0; i < 2; i++) bu[i] = *(const uint4*)(Brow + k0 + i * 8);
        (void)buf;
#endif
    };
    auto store_tiles = [&](int buf) {
#ifndef HAVE_ASYNC_LDS
        #pragma unroll
        for (int i = 0; i < 4; i++)
            *(uint4*)&sA[buf][t * 32 + i * 8] = au[i];
        #pragma unroll
        for (int i = 0; i < 2; i++)
            *(uint4*)&sB[buf][(t >> 1) * 32 + (t & 1) * 16 + i * 8] = bu[i];
#else
        (void)buf;
#endif
    };

    load_tiles(0, 0);
    store_tiles(0);
#ifdef HAVE_ASYNC_LDS
    wait_async0();
#endif
    __syncthreads();

    for (int k0 = 0; k0 < 1024; k0 += 32) {
        const int cur = (k0 >> 5) & 1;
        const bool more = (k0 + 32) < 1024;
        if (more) load_tiles(k0 + 32, cur ^ 1);

        const v16bf a0 = *(const v16bf*)&sA[cur][(wave * 32 + ln) * 32 + hi * 16];
        const v16bf a1 = *(const v16bf*)&sA[cur][(wave * 32 + 16 + ln) * 32 + hi * 16];
        const v16bf b0 = *(const v16bf*)&sB[cur][(0 * 16 + ln) * 32 + hi * 16];
        const v16bf b1 = *(const v16bf*)&sB[cur][(1 * 16 + ln) * 32 + hi * 16];
        const v16bf b2 = *(const v16bf*)&sB[cur][(2 * 16 + ln) * 32 + hi * 16];
        const v16bf b3 = *(const v16bf*)&sB[cur][(3 * 16 + ln) * 32 + hi * 16];
        acc[0][0] = WMMA_BF16(a0, b0, acc[0][0]);
        acc[1][0] = WMMA_BF16(a1, b0, acc[1][0]);
        acc[0][1] = WMMA_BF16(a0, b1, acc[0][1]);
        acc[1][1] = WMMA_BF16(a1, b1, acc[1][1]);
        acc[0][2] = WMMA_BF16(a0, b2, acc[0][2]);
        acc[1][2] = WMMA_BF16(a1, b2, acc[1][2]);
        acc[0][3] = WMMA_BF16(a0, b3, acc[0][3]);
        acc[1][3] = WMMA_BF16(a1, b3, acc[1][3]);

        if (more) store_tiles(cur ^ 1);
#ifdef HAVE_ASYNC_LDS
        if (more) wait_async0();
#endif
        __syncthreads();
    }

    #pragma unroll
    for (int mt = 0; mt < 2; mt++) {
        #pragma unroll
        for (int nt = 0; nt < 4; nt++) {
            const int col = n0 + nt * 16 + ln;
            #pragma unroll
            for (int r = 0; r < 8; r++) {
                const int m = m0 + wave * 32 + mt * 16 + r + hi * 8;
                out[(size_t)m * 1024 + col] = acc[mt][nt][r];
            }
        }
    }
}

// ---------------------------------------------------------------------------
// Host launch. Inputs: hidden_states(f32), positions(i32), causal_mask(bool,
// unused — causality computed analytically), Wqkv(f32), Wout(f32).
// ---------------------------------------------------------------------------
extern "C" void kernel_launch(void* const* d_in, const int* in_sizes, int n_in,
                              void* d_out, int out_size, void* d_ws, size_t ws_size,
                              hipStream_t stream)
{
    const float* X    = (const float*)d_in[0];
    const int*   pos  = (const int*)d_in[1];
    const float* Wqkv = (const float*)d_in[3];
    const float* Wout = (const float*)d_in[4];
    float* out = (float*)d_out;

    uint16_t* ws = (uint16_t*)d_ws;
    const size_t T = (size_t)4 * 16 * 1024 * 64;   // 4,194,304 elems (8 MB bf16)
    uint16_t* qb    = ws;                           // q  [B*H, S, Dh]
    uint16_t* kbuf  = ws + T;                       // k  [B*H, S, Dh]
    uint16_t* vT    = ws + 2 * T;                   // v  [B*H, Dh, S] (transposed)
    uint16_t* ab    = ws + 3 * T;                   // attn output [B*S, D] bf16
    uint16_t* Xb    = ws + 4 * T;                   // bf16 X      (4,194,304)
    uint16_t* Wqb   = ws + 5 * T;                   // bf16 Wqkv   (3,145,728)
    uint16_t* Woutb = ws + 5 * T + (size_t)3145728; // bf16 Wout   (1,048,576)

    // Streaming fp32->bf16 conversions (counts in float4 units).
    cvt_bf16_kernel<<<4096, 256, 0, stream>>>(X,    Xb,    1048576);
    cvt_bf16_kernel<<<3072, 256, 0, stream>>>(Wqkv, Wqb,    786432);
    cvt_bf16_kernel<<<1024, 256, 0, stream>>>(Wout, Woutb,  262144);

    qkv_rope_kernel<<<dim3(48, 32), 128, 0, stream>>>(Xb, Wqb, pos, qb, kbuf, vT);
    attn_kernel<<<dim3(4096), 32, 0, stream>>>(qb, kbuf, vT, ab);
    out_gemm_kernel<<<dim3(16, 32), 128, 0, stream>>>(ab, Woutb, out);
}
